// TransformerBlockQuantum_65481071395389
// MI455X (gfx1250) — matrix operand order repaired and checked by reading, not compiled, not run on hardware
//
#include <hip/hip_runtime.h>

typedef __attribute__((ext_vector_type(16))) __bf16 v16bf;
typedef __attribute__((ext_vector_type(8)))  float  v8f;

#define B_    2
#define S_    2048
#define E_    768
#define H_    12
#define DK_   64
#define NQ_   8
#define FF_   3072
#define ROWS_ (B_ * S_)   // 4096

// ---- WMMA operand index helpers (ISA 7.12.2, wave32) ----
// A 16x32 bf16: lane<16 row M=lane, K in {0..7,16..23}; lane>=16 same M, K in {8..15,24..31}
__device__ __forceinline__ int a_klocal(int e, int half) {
  int v = e >> 1, sl = e & 1;
  return (v < 4) ? (half * 8 + 2 * v + sl) : (16 + half * 8 + 2 * (v - 4) + sl);
}
// B 32x16 bf16: lane<16 col N=lane, K in {0..7,16..23}; lane>=16 same N, K in {8..15,24..31}
__device__ __forceinline__ int b_klocal(int e, int half) {
  int v = e >> 1, sl = e & 1;
  return (v < 4) ? (half * 8 + 2 * v + sl) : (16 + half * 8 + 2 * (v - 4) + sl);
}

// ---- DPP16 rotate-reduce within each 16-lane half (no LDS traffic) ----
template <int CTRL>
__device__ __forceinline__ float dpp_rot(float x) {
  int i = __builtin_amdgcn_update_dpp(0, __builtin_bit_cast(int, x), CTRL, 0xf, 0xf, true);
  return __builtin_bit_cast(float, i);
}
__device__ __forceinline__ float half_max(float v) {
  v = fmaxf(v, dpp_rot<0x121>(v));   // row_ror:1
  v = fmaxf(v, dpp_rot<0x122>(v));   // row_ror:2
  v = fmaxf(v, dpp_rot<0x124>(v));   // row_ror:4
  v = fmaxf(v, dpp_rot<0x128>(v));   // row_ror:8
  return v;
}
__device__ __forceinline__ float half_sum(float v) {
  v += dpp_rot<0x121>(v);
  v += dpp_rot<0x122>(v);
  v += dpp_rot<0x124>(v);
  v += dpp_rot<0x128>(v);
  return v;
}

struct alignas(16) Pack8 { __bf16 h[8]; };   // one ds_store_b128

// =====================================================================
// Packers: gather f32 -> fragment-native bf16
// =====================================================================

// x -> K^T B-fragments: frag id = ((b*H+h)*(S/16)+keytile)*2 + c   (c: feature chunk of 32)
__global__ __launch_bounds__(128)
void pack_kt_kernel(const float* __restrict__ x, __bf16* __restrict__ out)
{
  const int lane = threadIdx.x & 31, w = threadIdx.x >> 5;
  const int frag = blockIdx.x * 4 + w;
  const int half = lane >> 4, l16 = lane & 15;
  const int c  = frag & 1;
  const int kt = (frag >> 1) % (S_ / 16);
  const int bh = (frag >> 1) / (S_ / 16);
  const int b = bh / H_, h = bh % H_;
  const float* xb = x + (size_t)b * S_ * E_ + h * DK_;
  v16bf v;
#pragma unroll
  for (int e = 0; e < 16; ++e)
    v[e] = (__bf16)xb[(size_t)(kt * 16 + l16) * E_ + c * 32 + b_klocal(e, half)];
  ((v16bf*)out)[(size_t)frag * 32 + lane] = v;
}

// x -> V B-fragments: frag id = ((b*H+h)*(S/32)+keychunk)*4 + f   (f: feature tile of 16)
__global__ __launch_bounds__(128)
void pack_v_kernel(const float* __restrict__ x, __bf16* __restrict__ out)
{
  const int lane = threadIdx.x & 31, w = threadIdx.x >> 5;
  const int frag = blockIdx.x * 4 + w;
  const int half = lane >> 4, l16 = lane & 15;
  const int f  = frag & 3;
  const int jc = (frag >> 2) % (S_ / 32);
  const int bh = (frag >> 2) / (S_ / 32);
  const int b = bh / H_, h = bh % H_;
  const float* xb = x + (size_t)b * S_ * E_ + h * DK_;
  v16bf v;
#pragma unroll
  for (int e = 0; e < 16; ++e)
    v[e] = (__bf16)xb[(size_t)(jc * 32 + b_klocal(e, half)) * E_ + f * 16 + l16];
  ((v16bf*)out)[(size_t)frag * 32 + lane] = v;
}

// W[K,E] -> B-fragments: frag id = kc*(E/16) + ntile
__global__ __launch_bounds__(128)
void pack_w_kernel(const float* __restrict__ W, __bf16* __restrict__ out, int K)
{
  const int lane = threadIdx.x & 31, w = threadIdx.x >> 5;
  const int frag = blockIdx.x * 4 + w;
  const int half = lane >> 4, l16 = lane & 15;
  const int NT = E_ / 16;
  const int kc = frag / NT, nt = frag % NT;
  v16bf v;
#pragma unroll
  for (int e = 0; e < 16; ++e)
    v[e] = (__bf16)W[(size_t)(kc * 32 + b_klocal(e, half)) * E_ + nt * 16 + l16];
  ((v16bf*)out)[(size_t)frag * 32 + lane] = v;
}

// =====================================================================
// Flash attention (q = k = v = x per head), 1 wave = 32 query rows.
// 1/sqrt(DK)=1/8 is folded into the Q fragments (exact power-of-two scale).
// =====================================================================
__global__ __launch_bounds__(128)
void attn_kernel(const float* __restrict__ x,
                 const __bf16* __restrict__ xKT, const __bf16* __restrict__ xV,
                 __bf16* __restrict__ attnP)
{
  const int lane = threadIdx.x & 31;
  const int w    = threadIdx.x >> 5;
  const int wid  = blockIdx.x * 4 + w;

  const int b  = wid / (H_ * (S_ / 32));
  const int h  = (wid / (S_ / 32)) % H_;
  const int i0 = (wid % (S_ / 32)) * 32;

  const int half = lane >> 4;
  const int l16  = lane & 15;

  const float* xb = x + (size_t)b * S_ * E_ + h * DK_;
  const v16bf* ktv = (const v16bf*)xKT;
  const v16bf* vv  = (const v16bf*)xV;
  const size_t ktbase = (size_t)(b * H_ + h) * (S_ / 16);
  const size_t vbase  = (size_t)(b * H_ + h) * (S_ / 32);

  // per-wave staging, stored transposed [col][row] so repack stores are b128
  __shared__ __align__(32) __bf16 pbufT[4][2][32][16];

  // Q fragments: 2 row tiles x 2 K-chunks of 32, pre-scaled by 0.125
  v16bf qa[2][2];
#pragma unroll
  for (int ti = 0; ti < 2; ++ti)
#pragma unroll
    for (int c = 0; c < 2; ++c)
#pragma unroll
      for (int e = 0; e < 16; ++e)
        qa[ti][c][e] = (__bf16)(0.125f *
            xb[(size_t)(i0 + ti * 16 + l16) * E_ + c * 32 + a_klocal(e, half)]);

  v8f   o[2][4];
  float mrow[2][8], lrow[2][8];
#pragma unroll
  for (int ti = 0; ti < 2; ++ti) {
#pragma unroll
    for (int f = 0; f < 4; ++f) o[ti][f] = (v8f){};
#pragma unroll
    for (int r = 0; r < 8; ++r) { mrow[ti][r] = -1e30f; lrow[ti][r] = 0.f; }
  }

  for (int j0 = 0; j0 < S_; j0 += 32) {
    // pre-packed K^T and V fragments: one v16bf load each
    v16bf kb[2][2], vb[4];
#pragma unroll
    for (int s = 0; s < 2; ++s)
#pragma unroll
      for (int c = 0; c < 2; ++c)
        kb[s][c] = ktv[((ktbase + (j0 >> 4) + s) * 2 + c) * 32 + lane];
#pragma unroll
    for (int f = 0; f < 4; ++f)
      vb[f] = vv[((vbase + (j0 >> 5)) * 4 + f) * 32 + lane];
    if (j0 + 32 < S_) {   // prefetch next chunk
      __builtin_prefetch(&ktv[((ktbase + ((j0 + 32) >> 4)) * 2) * 32 + lane], 0, 0);
      __builtin_prefetch(&vv[((vbase + ((j0 + 32) >> 5)) * 4) * 32 + lane], 0, 0);
    }

    // scores = (Q/8) K^T
    v8f sc[2][2];
#pragma unroll
    for (int ti = 0; ti < 2; ++ti)
#pragma unroll
      for (int s = 0; s < 2; ++s) {
        v8f acc = (v8f){};
        acc = __builtin_amdgcn_wmma_f32_16x16x32_bf16(false, qa[ti][0], false, kb[s][0], (short)0, acc, false, false);
        acc = __builtin_amdgcn_wmma_f32_16x16x32_bf16(false, qa[ti][1], false, kb[s][1], (short)0, acc, false, false);
        sc[ti][s] = acc;
      }

    // online softmax; row reductions via DPP16 rotate-reduce (16-lane halves)
#pragma unroll
    for (int ti = 0; ti < 2; ++ti)
#pragma unroll
      for (int r = 0; r < 8; ++r) {
        float mv = half_max(fmaxf(sc[ti][0][r], sc[ti][1][r]));
        float mnew  = fmaxf(mrow[ti][r], mv);
        float alpha = __expf(mrow[ti][r] - mnew);
        mrow[ti][r] = mnew;
        float p0 = __expf(sc[ti][0][r] - mnew);
        float p1 = __expf(sc[ti][1][r] - mnew);
        float rs = half_sum(p0 + p1);
        lrow[ti][r] = lrow[ti][r] * alpha + rs;
#pragma unroll
        for (int f = 0; f < 4; ++f) o[ti][f][r] *= alpha;
        sc[ti][0][r] = p0; sc[ti][1][r] = p1;
      }

    // repack P (C-layout) -> A-layout via transposed LDS staging (b128 stores)
#pragma unroll
    for (int ti = 0; ti < 2; ++ti)
#pragma unroll
      for (int s = 0; s < 2; ++s) {
        Pack8 pk;
#pragma unroll
        for (int r = 0; r < 8; ++r) pk.h[r] = (__bf16)sc[ti][s][r];
        *(Pack8*)&pbufT[w][ti][s * 16 + l16][half * 8] = pk;
      }
    asm volatile("s_wait_dscnt 0" ::: "memory");
    v16bf pa[2];
#pragma unroll
    for (int ti = 0; ti < 2; ++ti)
#pragma unroll
      for (int e = 0; e < 16; ++e)
        pa[ti][e] = pbufT[w][ti][a_klocal(e, half)][l16];

#pragma unroll
    for (int ti = 0; ti < 2; ++ti)
#pragma unroll
      for (int f = 0; f < 4; ++f)
        o[ti][f] = __builtin_amdgcn_wmma_f32_16x16x32_bf16(false, pa[ti], false, vb[f], (short)0, o[ti][f], false, false);
  }

  // epilogue: normalize, repack O -> A-fragment bf16 via transposed LDS, store packed
  __bf16 (*obufT)[16] = (__bf16(*)[16])&pbufT[w][0][0][0];   // 64 cols x 16 rows view
  const int KC1 = E_ / 32;   // 24 K-chunks in the W_o GEMM
#pragma unroll
  for (int ti = 0; ti < 2; ++ti) {
    float inv[8];
#pragma unroll
    for (int r = 0; r < 8; ++r) inv[r] = 1.f / lrow[ti][r];
#pragma unroll
    for (int f = 0; f < 4; ++f) {
      Pack8 pk;
#pragma unroll
      for (int r = 0; r < 8; ++r) pk.h[r] = (__bf16)(o[ti][f][r] * inv[r]);
      *(Pack8*)&obufT[f * 16 + l16][half * 8] = pk;
    }
    asm volatile("s_wait_dscnt 0" ::: "memory");
    const int mt = (b * S_ + i0) / 16 + ti;   // global 16-row tile index
#pragma unroll
    for (int c = 0; c < 2; ++c) {
      v16bf av;
#pragma unroll
      for (int e = 0; e < 16; ++e)
        av[e] = obufT[c * 32 + a_klocal(e, half)][l16];
      ((v16bf*)attnP)[(size_t)(mt * KC1 + h * 2 + c) * 32 + lane] = av;
    }
    asm volatile("s_wait_dscnt 0" ::: "memory");
  }
}

// =====================================================================
// 64x64-tile WMMA GEMM on pre-packed bf16 fragments, fused residual+bias
// =====================================================================
__global__ __launch_bounds__(128)
void gemm_kernel(const __bf16* __restrict__ Ap, const __bf16* __restrict__ Bp,
                 const float* __restrict__ res, const float* __restrict__ bias,
                 float* __restrict__ C, int K)
{
  const int lane = threadIdx.x & 31;
  const int w    = threadIdx.x >> 5;
  const int t    = blockIdx.x * 4 + w;
  const int NT   = E_ / 64;             // 12
  const int m0   = (t / NT) * 64;
  const int n0   = (t % NT) * 64;
  const int half = lane >> 4, l16 = lane & 15;
  const int KC   = K / 32;
  const v16bf* Av = (const v16bf*)Ap;
  const v16bf* Bv = (const v16bf*)Bp;

  v8f acc[4][4];
#pragma unroll
  for (int i = 0; i < 4; ++i)
#pragma unroll
    for (int j = 0; j < 4; ++j) acc[i][j] = (v8f){};

  for (int kc = 0; kc < KC; ++kc) {
    v16bf a[4], bfr[4];
#pragma unroll
    for (int i = 0; i < 4; ++i)
      a[i] = Av[(size_t)((m0 / 16 + i) * KC + kc) * 32 + lane];
#pragma unroll
    for (int j = 0; j < 4; ++j)
      bfr[j] = Bv[(size_t)(kc * (E_ / 16) + n0 / 16 + j) * 32 + lane];
    if (kc + 1 < KC) {
      __builtin_prefetch(&Av[(size_t)((m0 / 16) * KC + kc + 1) * 32 + lane], 0, 0);
      __builtin_prefetch(&Bv[(size_t)((kc + 1) * (E_ / 16) + n0 / 16) * 32 + lane], 0, 0);
    }
#pragma unroll
    for (int i = 0; i < 4; ++i)
#pragma unroll
      for (int j = 0; j < 4; ++j)
        acc[i][j] = __builtin_amdgcn_wmma_f32_16x16x32_bf16(false, a[i], false, bfr[j], (short)0, acc[i][j], false, false);
  }

#pragma unroll
  for (int i = 0; i < 4; ++i)
#pragma unroll
    for (int r = 0; r < 8; ++r) {
      int row = m0 + i * 16 + half * 8 + r;
#pragma unroll
      for (int j = 0; j < 4; ++j) {
        int col = n0 + j * 16 + l16;
        C[(size_t)row * E_ + col] = acc[i][j][r] + res[(size_t)row * E_ + col] + bias[col];
      }
    }
}

// =====================================================================
// LayerNorm over rows of 768; optionally emits qm = cos(y[:8])*cos(ry)
// =====================================================================
__global__ __launch_bounds__(256)
void ln_kernel(const float* __restrict__ in, const float* __restrict__ g,
               const float* __restrict__ be, float* __restrict__ out,
               float* __restrict__ qm, const float* __restrict__ ry)
{
  const int row = blockIdx.x;
  const int tid = threadIdx.x;
  const float* rp = in + (size_t)row * E_;
  float v0 = rp[tid], v1 = rp[tid + 256], v2 = rp[tid + 512];
  float s = v0 + v1 + v2, ss = v0 * v0 + v1 * v1 + v2 * v2;
  __shared__ float rs[256], rss[256];
  rs[tid] = s; rss[tid] = ss;
  __syncthreads();
  for (int o = 128; o > 0; o >>= 1) {
    if (tid < o) { rs[tid] += rs[tid + o]; rss[tid] += rss[tid + o]; }
    __syncthreads();
  }
  float mean = rs[0] * (1.f / E_);
  float var  = rss[0] * (1.f / E_) - mean * mean;
  float rstd = rsqrtf(var + 1e-5f);
  float y0 = (v0 - mean) * rstd * g[tid] + be[tid];
  out[(size_t)row * E_ + tid]       = y0;
  out[(size_t)row * E_ + tid + 256] = (v1 - mean) * rstd * g[tid + 256] + be[tid + 256];
  out[(size_t)row * E_ + tid + 512] = (v2 - mean) * rstd * g[tid + 512] + be[tid + 512];
  if (qm && tid < NQ_) qm[row * NQ_ + tid] = cosf(y0) * cosf(ry[tid]);
}

// =====================================================================
// FFN stage 1 (K = 8, VALU): one wave per A-fragment of h1,
// writes h1 = relu(qm @ W1 + b1) directly in A-packed bf16.
// =====================================================================
__global__ __launch_bounds__(128)
void ffn1_kernel(const float* __restrict__ qm, const float* __restrict__ W1,
                 const float* __restrict__ b1, __bf16* __restrict__ h1P)
{
  const int lane = threadIdx.x & 31, w = threadIdx.x >> 5;
  const int frag = blockIdx.x * 4 + w;
  const int half = lane >> 4, l16 = lane & 15;
  const int KC = FF_ / 32;                 // 96
  const int mt = frag / KC, kc = frag % KC;
  const int row = mt * 16 + l16;

  float q[NQ_];
#pragma unroll
  for (int i = 0; i < NQ_; ++i) q[i] = qm[row * NQ_ + i];

  v16bf v;
#pragma unroll
  for (int e = 0; e < 16; ++e) {
    int f = kc * 32 + a_klocal(e, half);
    float acc = b1[f];
#pragma unroll
    for (int i = 0; i < NQ_; ++i) acc += q[i] * W1[i * FF_ + f];
    v[e] = (__bf16)fmaxf(acc, 0.f);
  }
  ((v16bf*)h1P)[(size_t)frag * 32 + lane] = v;
}

extern "C" void kernel_launch(void* const* d_in, const int* in_sizes, int n_in,
                              void* d_out, int out_size, void* d_ws, size_t ws_size,
                              hipStream_t stream)
{
  const float* x   = (const float*)d_in[0];
  const float* W_o = (const float*)d_in[1];
  const float* b_o = (const float*)d_in[2];
  const float* g1  = (const float*)d_in[3];
  const float* be1 = (const float*)d_in[4];
  const float* g2  = (const float*)d_in[5];
  const float* be2 = (const float*)d_in[6];
  const float* ry  = (const float*)d_in[7];
  const float* W1  = (const float*)d_in[8];
  const float* b1  = (const float*)d_in[9];
  const float* W2  = (const float*)d_in[10];
  const float* b2  = (const float*)d_in[11];

  // 256B-aligned workspace carve-up
  size_t off = 0;
  auto carve = [&](size_t bytes) -> void* {
    off = (off + 255) & ~(size_t)255;
    void* p = (char*)d_ws + off;
    off += bytes;
    return p;
  };
  float*  t1   = (float*) carve((size_t)ROWS_ * E_ * 4);
  float*  t2   = (float*) carve((size_t)ROWS_ * E_ * 4);
  float*  x1   = (float*) carve((size_t)ROWS_ * E_ * 4);
  float*  qm   = (float*) carve((size_t)ROWS_ * NQ_ * 4);
  __bf16* xKT  = (__bf16*)carve((size_t)B_ * S_ * E_ * 2);
  __bf16* xV   = (__bf16*)carve((size_t)B_ * S_ * E_ * 2);
  __bf16* attnP= (__bf16*)carve((size_t)ROWS_ * E_ * 2);
  __bf16* WoP  = (__bf16*)carve((size_t)E_ * E_ * 2);
  __bf16* W2P  = (__bf16*)carve((size_t)FF_ * E_ * 2);
  __bf16* h1P  = (__bf16*)carve((size_t)ROWS_ * FF_ * 2);

  // operand packing (one cheap pass each)
  pack_kt_kernel<<<(B_ * H_ * (S_ / 16) * 2) / 4, 128, 0, stream>>>(x, xKT);
  pack_v_kernel <<<(B_ * H_ * (S_ / 32) * 4) / 4, 128, 0, stream>>>(x, xV);
  pack_w_kernel <<<((E_  / 32) * (E_ / 16)) / 4, 128, 0, stream>>>(W_o, WoP, E_);
  pack_w_kernel <<<((FF_ / 32) * (E_ / 16)) / 4, 128, 0, stream>>>(W2, W2P, FF_);

  attn_kernel<<<(B_ * H_ * (S_ / 32)) / 4, 128, 0, stream>>>(x, xKT, xV, attnP);
  gemm_kernel<<<((ROWS_ / 64) * (E_ / 64)) / 4, 128, 0, stream>>>(attnP, WoP, x, b_o, t1, E_);
  ln_kernel<<<ROWS_, 256, 0, stream>>>(t1, g1, be1, x1, qm, ry);
  ffn1_kernel<<<((ROWS_ / 16) * (FF_ / 32)) / 4, 128, 0, stream>>>(qm, W1, b1, h1P);
  gemm_kernel<<<((ROWS_ / 64) * (E_ / 64)) / 4, 128, 0, stream>>>(h1P, W2P, x1, b2, t2, FF_);
  ln_kernel<<<ROWS_, 256, 0, stream>>>(t2, g2, be2, (float*)d_out, nullptr, nullptr);
}
